// DETR_1846835937928
// MI455X (gfx1250) — matrix operand order repaired and checked by reading, not compiled, not run on hardware
//
#include <hip/hip_runtime.h>

typedef __attribute__((ext_vector_type(16))) _Float16 v16h;
typedef __attribute__((ext_vector_type(8)))  _Float16 v8h;
typedef __attribute__((ext_vector_type(8)))  float    v8f;

#define GF_BIAS   1
#define GF_RESID  2
#define GF_GELU   4
#define GF_STOREH 8

union FragU { v16h v; v8h h[2]; };

// One 16x16x32 WMMA step from the shared tiles (A: m-major, B: n-major, stride 40 halfs).
__device__ __forceinline__ v8f wmma_tile(const _Float16* As, const _Float16* Bs,
                                         int wm, int wn, int lm, int hf, v8f acc)
{
    FragU fa, fb;
    fa.h[0] = *(const v8h*)&As[(wm + lm) * 40 + hf * 8];
    fa.h[1] = *(const v8h*)&As[(wm + lm) * 40 + 16 + hf * 8];
    fb.h[0] = *(const v8h*)&Bs[(wn + lm) * 40 + hf * 8];
    fb.h[1] = *(const v8h*)&Bs[(wn + lm) * 40 + 16 + hf * 8];
    return __builtin_amdgcn_wmma_f32_16x16x32_f16(false, fa.v, false, fb.v,
                                                  (short)0, acc, false, false);
}

__device__ __forceinline__ v8h pack8(float4 x0, float4 x1)
{
    v8h ah;
    ah[0] = (_Float16)x0.x; ah[1] = (_Float16)x0.y;
    ah[2] = (_Float16)x0.z; ah[3] = (_Float16)x0.w;
    ah[4] = (_Float16)x1.x; ah[5] = (_Float16)x1.y;
    ah[6] = (_Float16)x1.z; ah[7] = (_Float16)x1.w;
    return ah;
}

// ---------------------------------------------------------------------------
// Generic batched WMMA GEMM:  C[b] = act( alpha * A[b] * B[b]^T-layout + bias )
//   A: f32 (sAm, sAk), converted to f16 on the fly into LDS.
//   Bt: f16, element (n,k) at n*sBn + k*sBk (repacked weights or f16 acts).
//   batch = blockIdx.y decomposed as outer*batchInner + inner.
//   K-loop specialized (vec/vec, vec/gather, gather/gather) with the uniform
//   selection hoisted out of the loop; pointers advance by increments only.
// ---------------------------------------------------------------------------
__global__ __launch_bounds__(128) void wmma_gemm_k(
    const float* __restrict__ A, const _Float16* __restrict__ Bt,
    const float* __restrict__ bias, const float* __restrict__ resid,
    float* __restrict__ C, _Float16* __restrict__ Ch,
    int M, int N, int K,
    int sAm, int sAk, int sBn, int sBk, int sCm,
    int tilesN, int batchInner,
    int sAo, int sAi, int sBo, int sBi, long sCo, long sCi,
    float alpha, int flags, int residMod)
{
    __shared__ __align__(16) _Float16 As[32 * 40];
    __shared__ __align__(16) _Float16 Bs[32 * 40];

    int tid  = threadIdx.x;
    int tm   = (int)blockIdx.x / tilesN;
    int tn   = (int)blockIdx.x % tilesN;
    int bidx = (int)blockIdx.y;
    int bo   = bidx / batchInner;
    int bi   = bidx - bo * batchInner;

    const float*    Ab = A  + (long)(bo * sAo + bi * sAi);
    const _Float16* Bb = Bt + (long)(bo * sBo + bi * sBi);
    long coff = (long)bo * sCo + (long)bi * sCi;

    int lr = tid >> 2;           // 0..31 : row within tile (A: m, B: n)
    int kg = (tid & 3) * 8;      // k sub-offset

    int wave = tid >> 5;
    int lane = tid & 31;
    int wm = (wave >> 1) * 16, wn = (wave & 1) * 16;
    int lm = lane & 15, hf = lane >> 4;

    // clamped row pointers (garbage rows/cols are masked at the epilogue)
    int gmc = tm * 32 + lr; if (gmc > M - 1) gmc = M - 1;
    int gnc = tn * 32 + lr; if (gnc > N - 1) gnc = N - 1;
    const float*    aRow = Ab + (long)gmc * sAm;
    const _Float16* bRow = Bb + (long)gnc * sBn;
    _Float16* asl = &As[lr * 40 + kg];
    _Float16* bsl = &Bs[lr * 40 + kg];

    // uniform vectorization predicates (16B-aligned contiguous k)
    bool avec = (sAk == 1) && (((sAm | sAo | sAi) & 3) == 0);
    bool bvec = (sBk == 1) && (((sBn | sBo | sBi) & 7) == 0);

    int kFull = K & ~31;
    v8f acc = {};

    if (avec && bvec) {
        const float4* ap = (const float4*)(aRow + kg);
        const v8h*    bp = (const v8h*)(bRow + kg);
        for (int k0 = 0; k0 < kFull; k0 += 32, ap += 8, bp += 4) {
            float4 x0 = ap[0], x1 = ap[1];
            *(v8h*)asl = pack8(x0, x1);
            *(v8h*)bsl = bp[0];
            __syncthreads();
            acc = wmma_tile(As, Bs, wm, wn, lm, hf, acc);
            __syncthreads();
        }
    } else if (avec) {
        const float4*   ap = (const float4*)(aRow + kg);
        const _Float16* bp = bRow + (long)kg * sBk;
        long bstep = 32L * sBk;
        for (int k0 = 0; k0 < kFull; k0 += 32, ap += 8, bp += bstep) {
            float4 x0 = ap[0], x1 = ap[1];
            *(v8h*)asl = pack8(x0, x1);
            v8h bh;
#pragma unroll
            for (int j = 0; j < 8; ++j) bh[j] = bp[(long)j * sBk];
            *(v8h*)bsl = bh;
            __syncthreads();
            acc = wmma_tile(As, Bs, wm, wn, lm, hf, acc);
            __syncthreads();
        }
    } else {
        const float*    ap = aRow + (long)kg * sAk;
        const _Float16* bp = bRow + (long)kg * sBk;
        long astep = 32L * sAk, bstep = 32L * sBk;
        for (int k0 = 0; k0 < kFull; k0 += 32, ap += astep, bp += bstep) {
            v8h ah, bh;
#pragma unroll
            for (int j = 0; j < 8; ++j) {
                ah[j] = (_Float16)ap[(long)j * sAk];
                bh[j] = bp[(long)j * sBk];
            }
            *(v8h*)asl = ah;
            *(v8h*)bsl = bh;
            __syncthreads();
            acc = wmma_tile(As, Bs, wm, wn, lm, hf, acc);
            __syncthreads();
        }
    }

    if (kFull < K) {                         // K % 32 tail (clamped selects, no exec churn)
        int ka = kFull + kg;
        v8h ah, bh;
#pragma unroll
        for (int j = 0; j < 8; ++j) {
            int k = ka + j;
            int kc = (k < K) ? k : (K - 1);
            float va = aRow[(long)kc * sAk];
            _Float16 vb = bRow[(long)kc * sBk];
            ah[j] = (k < K) ? (_Float16)va : (_Float16)0.f;
            bh[j] = (k < K) ? vb : (_Float16)0.f;
        }
        *(v8h*)asl = ah;
        *(v8h*)bsl = bh;
        __syncthreads();
        acc = wmma_tile(As, Bs, wm, wn, lm, hf, acc);
        __syncthreads();
    }

    int gn = tn * 32 + wn + lm;
#pragma unroll
    for (int r = 0; r < 8; ++r) {
        int gm = tm * 32 + wm + r + 8 * hf;
        if (gm < M && gn < N) {
            float v = acc[r] * alpha;
            if (flags & GF_BIAS)  v += bias[gn];
            if (flags & GF_GELU)  v = 0.5f * v * (1.f + erff(v * 0.70710678118654752f));
            if (flags & GF_RESID) {
                long rrow = residMod ? (long)(gm % residMod) : (long)gm;
                v += resid[rrow * sCm + gn];
            }
            long ci = coff + (long)gm * sCm + gn;
            C[ci] = v;
            if (flags & GF_STOREH) Ch[ci] = (_Float16)v;
        }
    }
}

// ---------------------------------------------------------------------------
// Patch-embed conv as fused im2col WMMA GEMM.
// M = 6144 (b*192 + py*16 + px), N = 256, K = 16000 (c*1600 + r*40 + cc)
// 8-element k-groups never cross a 40-wide row (40 % 8 == 0) -> float4 loads.
// ---------------------------------------------------------------------------
__global__ __launch_bounds__(128) void conv_gemm_k(
    const float* __restrict__ x, const _Float16* __restrict__ Wt,
    const float* __restrict__ bias, const float* __restrict__ pos,
    float* __restrict__ C)
{
    __shared__ __align__(16) _Float16 As[32 * 40];
    __shared__ __align__(16) _Float16 Bs[32 * 40];
    const int K = 16000;

    int tid = threadIdx.x;
    int tm = (int)blockIdx.x >> 3;
    int tn = (int)blockIdx.x & 7;
    int lr = tid >> 2;
    int kg = (tid & 3) * 8;
    int wave = tid >> 5, lane = tid & 31;
    int wm = (wave >> 1) * 16, wn = (wave & 1) * 16;
    int lm = lane & 15, hf = lane >> 4;

    int gm = tm * 32 + lr;
    int bimg = gm / 192;
    int p = gm - bimg * 192;
    int py = p >> 4, px = p & 15;
    const float* xb = x + (long)bimg * 10 * 480 * 640 + (long)(py * 40) * 640 + px * 40;
    int gnl = tn * 32 + lr;
    const v8h* wp = (const v8h*)(Wt + (long)gnl * K + kg);
    _Float16* asl = &As[lr * 40 + kg];
    _Float16* bsl = &Bs[lr * 40 + kg];

    v8f acc = {};
    for (int k0 = 0; k0 < K; k0 += 32, wp += 4) {
        int k = k0 + kg;
        int c = k / 1600;
        int rem = k - c * 1600;
        int rr = rem / 40;
        int cc = rem - rr * 40;
        const float4* pv = (const float4*)(xb + ((long)c * 480 + rr) * 640 + cc);
        float4 x0 = pv[0], x1 = pv[1];
        *(v8h*)asl = pack8(x0, x1);
        *(v8h*)bsl = wp[0];

        __syncthreads();
        acc = wmma_tile(As, Bs, wm, wn, lm, hf, acc);
        __syncthreads();
    }

    int gn = tn * 32 + wn + lm;
#pragma unroll
    for (int r = 0; r < 8; ++r) {
        int gmr = tm * 32 + wm + r + 8 * hf;
        int pp = gmr % 192;
        C[(long)gmr * 256 + gn] = acc[r] + bias[gn] + pos[(long)pp * 256 + gn];
    }
}

// ---------------------------------------------------------------------------
// LayerNorm over last dim (256), one wave per row. wave32.
// ---------------------------------------------------------------------------
__global__ void layernorm_k(const float* __restrict__ in, const float* __restrict__ gam,
                            const float* __restrict__ bet, float* __restrict__ out, int rows)
{
    int wid = ((int)blockIdx.x * (int)blockDim.x + (int)threadIdx.x) >> 5;
    int lane = threadIdx.x & 31;
    if (wid >= rows) return;
    const float* p = in + (long)wid * 256;
    float v[8];
    float s = 0.f;
#pragma unroll
    for (int j = 0; j < 8; ++j) { v[j] = p[lane + 32 * j]; s += v[j]; }
#pragma unroll
    for (int m = 16; m > 0; m >>= 1) s += __shfl_xor(s, m, 32);
    float mean = s * (1.f / 256.f);
    float q = 0.f;
#pragma unroll
    for (int j = 0; j < 8; ++j) { float d = v[j] - mean; q += d * d; }
#pragma unroll
    for (int m = 16; m > 0; m >>= 1) q += __shfl_xor(q, m, 32);
    float inv = rsqrtf(q * (1.f / 256.f) + 1e-5f);
    float* o = out + (long)wid * 256;
#pragma unroll
    for (int j = 0; j < 8; ++j) {
        int c = lane + 32 * j;
        o[c] = (v[j] - mean) * inv * gam[c] + bet[c];
    }
}

// softmax over the Q axis (column softmax) of x[batch][Q][S], in place
__global__ void softmax_col_k(float* __restrict__ x, int Q, int S)
{
    long base = (long)blockIdx.y * Q * S;
    int s = (int)blockIdx.x * (int)blockDim.x + (int)threadIdx.x;
    if (s >= S) return;
    float* p = x + base + s;
    float mx = -3.4e38f;
    for (int q = 0; q < Q; ++q) mx = fmaxf(mx, p[(long)q * S]);
    float sum = 0.f;
    for (int q = 0; q < Q; ++q) { float e = expf(p[(long)q * S] - mx); p[(long)q * S] = e; sum += e; }
    float inv = 1.f / sum;
    for (int q = 0; q < Q; ++q) p[(long)q * S] *= inv;
}

// softmax over the last axis of x[rows][S], in place
__global__ void softmax_row_k(float* __restrict__ x, int rows, int S)
{
    int r = (int)blockIdx.x * (int)blockDim.x + (int)threadIdx.x;
    if (r >= rows) return;
    float* p = x + (long)r * S;
    float mx = -3.4e38f;
    for (int i = 0; i < S; ++i) mx = fmaxf(mx, p[i]);
    float sum = 0.f;
    for (int i = 0; i < S; ++i) { float e = expf(p[i] - mx); p[i] = e; sum += e; }
    float inv = 1.f / sum;
    for (int i = 0; i < S; ++i) p[i] *= inv;
}

// cls/bbox heads: q (800,256) -> logits (800,2) then sigmoid(bbox) (800,4)
__global__ void heads_k(const float* __restrict__ q, const float* __restrict__ cw,
                        const float* __restrict__ cb, const float* __restrict__ bw,
                        const float* __restrict__ bb, float* __restrict__ out)
{
    int idx = (int)blockIdx.x * (int)blockDim.x + (int)threadIdx.x;
    if (idx >= 800 * 6) return;
    int row = idx / 6, j = idx - row * 6;
    const float* qr = q + (long)row * 256;
    float acc = 0.f;
    if (j < 2) {
        for (int e = 0; e < 256; ++e) acc += qr[e] * cw[e * 2 + j];
        out[row * 2 + j] = acc + cb[j];
    } else {
        int jj = j - 2;
        for (int e = 0; e < 256; ++e) acc += qr[e] * bw[e * 4 + jj];
        acc += bb[jj];
        out[1600 + row * 4 + jj] = 1.f / (1.f + expf(-acc));
    }
}

// out[c*R + r] = (f16) in[r*Cc + c], batched
__global__ void transpose_cvt_k(const float* __restrict__ in, _Float16* __restrict__ out,
                                int R, int Cc, long inB, long outB)
{
    long b = blockIdx.y;
    const float* ip = in + b * inB;
    _Float16* op = out + b * outB;
    int idx = (int)blockIdx.x * (int)blockDim.x + (int)threadIdx.x;
    if (idx >= R * Cc) return;
    int r = idx / Cc, c = idx - r * Cc;
    op[(long)c * R + r] = (_Float16)ip[(long)r * Cc + c];
}

__global__ void cvt_k(const float* __restrict__ in, _Float16* __restrict__ out, long n)
{
    long i = (long)blockIdx.x * 256 + threadIdx.x;
    if (i < n) out[i] = (_Float16)in[i];
}

// ---------------------------------------------------------------------------
extern "C" void kernel_launch(void* const* d_in, const int* in_sizes, int n_in,
                              void* d_out, int out_size, void* d_ws, size_t ws_size,
                              hipStream_t stream)
{
    (void)in_sizes; (void)n_in; (void)out_size; (void)ws_size;
    const float* X_IN   = (const float*)d_in[0];
    const float* CONV_W = (const float*)d_in[1];
    const float* CONV_B = (const float*)d_in[2];
    const float* POS    = (const float*)d_in[3];
    const float* QPOS   = (const float*)d_in[4];
    const int ENC0 = 5;              // 6 blocks x 15 leaves
    const int DEC0 = 5 + 6 * 15;     // 6 blocks x 24 leaves
    const int TAIL = DEC0 + 6 * 24;
    const float* CLS_W  = (const float*)d_in[TAIL + 0];
    const float* CLS_B  = (const float*)d_in[TAIL + 1];
    const float* BBOX_W = (const float*)d_in[TAIL + 2];
    const float* BBOX_B = (const float*)d_in[TAIL + 3];

    // ---- workspace bump allocator ----
    char* wsb = (char*)d_ws;
    size_t off = 0;
    auto alloc = [&](size_t bytes) -> void* {
        void* p = wsb + off;
        off = (off + bytes + 255) & ~(size_t)255;
        return p;
    };

    _Float16* convWT = (_Float16*)alloc(4096000ull * 2);
    struct EncW { _Float16 *wq, *wk, *wv, *w, *w1, *w2; } encw[6];
    struct DecW { _Float16 *swq, *swk, *swv, *sw, *cwq, *cwk, *cwv, *cw, *w1, *w2; } decw[6];
    for (int b = 0; b < 6; ++b) {
        encw[b].wq = (_Float16*)alloc(65536 * 2);
        encw[b].wk = (_Float16*)alloc(65536 * 2);
        encw[b].wv = (_Float16*)alloc(65536 * 2);
        encw[b].w  = (_Float16*)alloc(65536 * 2);
        encw[b].w1 = (_Float16*)alloc(262144 * 2);
        encw[b].w2 = (_Float16*)alloc(262144 * 2);
    }
    for (int b = 0; b < 6; ++b) {
        decw[b].swq = (_Float16*)alloc(65536 * 2);
        decw[b].swk = (_Float16*)alloc(65536 * 2);
        decw[b].swv = (_Float16*)alloc(65536 * 2);
        decw[b].sw  = (_Float16*)alloc(65536 * 2);
        decw[b].cwq = (_Float16*)alloc(65536 * 2);
        decw[b].cwk = (_Float16*)alloc(65536 * 2);
        decw[b].cwv = (_Float16*)alloc(65536 * 2);
        decw[b].cw  = (_Float16*)alloc(65536 * 2);
        decw[b].w1  = (_Float16*)alloc(262144 * 2);
        decw[b].w2  = (_Float16*)alloc(262144 * 2);
    }
    const long RSEQ = 6144;                       // encoder rows
    float* Xf  = (float*)alloc(RSEQ * 256 * 4);   // running activation / memory
    float* N1  = (float*)alloc(RSEQ * 256 * 4);   // LN output
    float* QB  = (float*)alloc(RSEQ * 256 * 4);   // Q f32
    float* OB  = (float*)alloc(RSEQ * 256 * 4);   // attn out; also K/V f32 dump
    float* ADD = (float*)alloc(RSEQ * 256 * 4);   // residual hub
    _Float16* KH = (_Float16*)alloc(RSEQ * 256 * 2);
    _Float16* VH = (_Float16*)alloc(RSEQ * 256 * 2);
    float* SC = (float*)alloc((size_t)256 * 192 * 192 * 4);  // scores; aliased as MLP hidden
    float* H1 = SC;                                          // lifetimes disjoint
    float* F  = (float*)alloc(800 * 256 * 4);
    float* F1 = (float*)alloc(800 * 256 * 4);
    float* DQ = (float*)alloc(800 * 256 * 4);

    // ---- helpers ----
    auto gemm = [&](const float* A, const _Float16* B, const float* bias, const float* resid,
                    float* C, _Float16* Ch, int M, int N, int K,
                    int sAm, int sAk, int sBn, int sBk, int sCm,
                    int batch, int bIn, int sAo, int sAi, int sBo, int sBi,
                    long sCo, long sCi, float alpha, int flags, int residMod) {
        int tm = (M + 31) / 32, tn = (N + 31) / 32;
        dim3 grid(tm * tn, batch);
        wmma_gemm_k<<<grid, 128, 0, stream>>>(A, B, bias, resid, C, Ch, M, N, K,
                                              sAm, sAk, sBn, sBk, sCm, tn, bIn,
                                              sAo, sAi, sBo, sBi, sCo, sCi,
                                              alpha, flags, residMod);
    };
    auto lnorm = [&](const float* in, const float* gam, const float* bet, float* out, int rows) {
        layernorm_k<<<dim3((rows + 7) / 8), 256, 0, stream>>>(in, gam, bet, out, rows);
    };
    auto tcvt = [&](const float* in, _Float16* out, int R, int Cc, int batch, long inB, long outB) {
        dim3 grid((R * Cc + 255) / 256, batch);
        transpose_cvt_k<<<grid, 256, 0, stream>>>(in, out, R, Cc, inB, outB);
    };
    auto F32 = [&](int i) -> const float* { return (const float*)d_in[i]; };

    // ---- weight repack ----
    cvt_k<<<dim3((4096000 + 255) / 256), 256, 0, stream>>>(CONV_W, convWT, 4096000);
    for (int b = 0; b < 6; ++b) {
        int e = ENC0 + b * 15;
        tcvt(F32(e + 0), encw[b].wq, 256, 32, 8, 8192, 8192);
        tcvt(F32(e + 2), encw[b].wk, 256, 32, 8, 8192, 8192);
        tcvt(F32(e + 4), encw[b].wv, 256, 32, 8, 8192, 8192);
        tcvt(F32(e + 6), encw[b].w, 256, 256, 1, 0, 0);
        tcvt(F32(e + 7), encw[b].w1, 256, 1024, 1, 0, 0);
        tcvt(F32(e + 9), encw[b].w2, 1024, 256, 1, 0, 0);
        int d = DEC0 + b * 24;
        tcvt(F32(d + 0),  decw[b].swq, 256, 32, 8, 8192, 8192);
        tcvt(F32(d + 2),  decw[b].swk, 256, 32, 8, 8192, 8192);
        tcvt(F32(d + 4),  decw[b].swv, 256, 32, 8, 8192, 8192);
        tcvt(F32(d + 6),  decw[b].sw, 256, 256, 1, 0, 0);
        tcvt(F32(d + 7),  decw[b].cwq, 256, 32, 8, 8192, 8192);
        tcvt(F32(d + 9),  decw[b].cwk, 256, 32, 8, 8192, 8192);
        tcvt(F32(d + 11), decw[b].cwv, 256, 32, 8, 8192, 8192);
        tcvt(F32(d + 13), decw[b].cw, 256, 256, 1, 0, 0);
        tcvt(F32(d + 14), decw[b].w1, 256, 1024, 1, 0, 0);
        tcvt(F32(d + 16), decw[b].w2, 1024, 256, 1, 0, 0);
    }

    // ---- patch embedding ----
    conv_gemm_k<<<dim3(192 * 8), 128, 0, stream>>>(X_IN, convWT, CONV_B, POS, Xf);

    const float RS = 0.17677669529663687f;   // 1/sqrt(32)

    // ---- encoder ----
    for (int b = 0; b < 6; ++b) {
        int e = ENC0 + b * 15;
        const float *bq = F32(e + 1), *bk = F32(e + 3), *bv = F32(e + 5);
        const float *b1 = F32(e + 8), *b2 = F32(e + 10);
        const float *g1 = F32(e + 11), *be1 = F32(e + 12), *g2 = F32(e + 13), *be2 = F32(e + 14);

        lnorm(Xf, g1, be1, N1, 6144);
        gemm(N1, encw[b].wq, bq, 0, QB, 0, 6144, 256, 256, 256, 1, 256, 1, 256,
             1, 1, 0, 0, 0, 0, 0, 0, 1.f, GF_BIAS, 0);
        gemm(N1, encw[b].wk, bk, 0, OB, KH, 6144, 256, 256, 256, 1, 256, 1, 256,
             1, 1, 0, 0, 0, 0, 0, 0, 1.f, GF_BIAS | GF_STOREH, 0);
        gemm(N1, encw[b].wv, bv, 0, OB, VH, 6144, 256, 256, 256, 1, 256, 1, 256,
             1, 1, 0, 0, 0, 0, 0, 0, 1.f, GF_BIAS | GF_STOREH, 0);
        // scores[b,h] = Q K^T / sqrt(d)
        gemm(QB, KH, 0, 0, SC, 0, 192, 192, 32, 256, 1, 256, 1, 192,
             256, 8, 192 * 256, 32, 192 * 256, 32, 8L * 192 * 192, 192L * 192, RS, 0, 0);
        softmax_col_k<<<dim3(3, 256), 64, 0, stream>>>(SC, 192, 192);   // axis=-2 (query)
        gemm(SC, VH, 0, 0, OB, 0, 192, 32, 192, 192, 1, 1, 256, 256,
             256, 8, 8 * 192 * 192, 192 * 192, 192 * 256, 32, 192L * 256, 32, 1.f, 0, 0);
        gemm(OB, encw[b].w, 0, Xf, ADD, 0, 6144, 256, 256, 256, 1, 256, 1, 256,
             1, 1, 0, 0, 0, 0, 0, 0, 1.f, GF_RESID, 0);
        lnorm(ADD, g2, be2, N1, 6144);
        gemm(N1, encw[b].w1, b1, 0, H1, 0, 6144, 1024, 256, 256, 1, 256, 1, 1024,
             1, 1, 0, 0, 0, 0, 0, 0, 1.f, GF_BIAS | GF_GELU, 0);
        gemm(H1, encw[b].w2, b2, ADD, Xf, 0, 6144, 256, 1024, 1024, 1, 1024, 1, 256,
             1, 1, 0, 0, 0, 0, 0, 0, 1.f, GF_BIAS | GF_RESID, 0);
    }

    // ---- decoder ----
    for (int b = 0; b < 6; ++b) {
        int d = DEC0 + b * 24;
        const float *sbq = F32(d + 1), *sbk = F32(d + 3), *sbv = F32(d + 5);
        const float *cbq = F32(d + 8), *cbk = F32(d + 10), *cbv = F32(d + 12);
        const float *mb1 = F32(d + 15), *mb2 = F32(d + 17);
        const float *g1 = F32(d + 18), *be1 = F32(d + 19), *g2 = F32(d + 20);
        const float *be2 = F32(d + 21), *g3 = F32(d + 22), *be3 = F32(d + 23);
        bool first = (b == 0);
        int Mq = first ? 25 : 800;
        const float* qact = first ? QPOS : DQ;
        int nb = first ? 8 : 256;
        int qo = first ? 0 : 25 * 256;

        // --- self attention ---
        gemm(qact, decw[b].swq, sbq, 0, QB, 0, Mq, 256, 256, 256, 1, 256, 1, 256,
             1, 1, 0, 0, 0, 0, 0, 0, 1.f, GF_BIAS, 0);
        gemm(qact, decw[b].swk, sbk, 0, OB, KH, Mq, 256, 256, 256, 1, 256, 1, 256,
             1, 1, 0, 0, 0, 0, 0, 0, 1.f, GF_BIAS | GF_STOREH, 0);
        gemm(qact, decw[b].swv, sbv, 0, OB, VH, Mq, 256, 256, 256, 1, 256, 1, 256,
             1, 1, 0, 0, 0, 0, 0, 0, 1.f, GF_BIAS | GF_STOREH, 0);
        gemm(QB, KH, 0, 0, SC, 0, 25, 25, 32, 256, 1, 256, 1, 25,
             nb, 8, qo, 32, qo, 32, 8L * 625, 625, RS, 0, 0);
        if (first)
            softmax_row_k<<<dim3(1), 256, 0, stream>>>(SC, 200, 25);      // 2D: key axis
        else
            softmax_col_k<<<dim3(1, 256), 64, 0, stream>>>(SC, 25, 25);   // 3D: query axis
        gemm(SC, VH, 0, 0, OB, 0, 25, 32, 25, 25, 1, 1, 256, 256,
             nb, 8, 8 * 625, 625, qo, 32, (long)qo, 32, 1.f, 0, 0);
        gemm(OB, decw[b].sw, 0, qact, F, 0, Mq, 256, 256, 256, 1, 256, 1, 256,
             1, 1, 0, 0, 0, 0, 0, 0, 1.f, GF_RESID, 0);
        lnorm(F, g1, be1, ADD, Mq);                 // ADD = q after LN1

        // --- cross attention (mem = Xf, 6144 rows) ---
        gemm(ADD, decw[b].cwq, cbq, 0, QB, 0, Mq, 256, 256, 256, 1, 256, 1, 256,
             1, 1, 0, 0, 0, 0, 0, 0, 1.f, GF_BIAS, 0);
        gemm(Xf, decw[b].cwk, cbk, 0, OB, KH, 6144, 256, 256, 256, 1, 256, 1, 256,
             1, 1, 0, 0, 0, 0, 0, 0, 1.f, GF_BIAS | GF_STOREH, 0);
        gemm(Xf, decw[b].cwv, cbv, 0, OB, VH, 6144, 256, 256, 256, 1, 256, 1, 256,
             1, 1, 0, 0, 0, 0, 0, 0, 1.f, GF_BIAS | GF_STOREH, 0);
        int caQo = first ? 0 : 25 * 256;            // Q broadcast over batch in block 0
        gemm(QB, KH, 0, 0, SC, 0, 25, 192, 32, 256, 1, 256, 1, 192,
             256, 8, caQo, 32, 192 * 256, 32, 8L * 25 * 192, 25L * 192, RS, 0, 0);
        softmax_col_k<<<dim3(3, 256), 64, 0, stream>>>(SC, 25, 192);      // query axis
        gemm(SC, VH, 0, 0, OB, 0, 25, 32, 192, 192, 1, 1, 256, 256,
             256, 8, 8 * 25 * 192, 25 * 192, 192 * 256, 32, 25L * 256, 32, 1.f, 0, 0);
        gemm(OB, decw[b].cw, 0, ADD, F, 0, 800, 256, 256, 256, 1, 256, 1, 256,
             1, 1, 0, 0, 0, 0, 0, 0, 1.f, GF_RESID, first ? 25 : 0);
        lnorm(F, g2, be2, F1, 800);

        // --- MLP ---
        gemm(F1, decw[b].w1, mb1, 0, H1, 0, 800, 1024, 256, 256, 1, 256, 1, 1024,
             1, 1, 0, 0, 0, 0, 0, 0, 1.f, GF_BIAS | GF_GELU, 0);
        gemm(H1, decw[b].w2, mb2, F1, F, 0, 800, 256, 1024, 1024, 1, 1024, 1, 256,
             1, 1, 0, 0, 0, 0, 0, 0, 1.f, GF_BIAS | GF_RESID, 0);
        lnorm(F, g3, be3, DQ, 800);
    }

    // ---- heads ----
    heads_k<<<dim3((4800 + 255) / 256), 256, 0, stream>>>(DQ, CLS_W, CLS_B,
                                                          BBOX_W, BBOX_B, (float*)d_out);
}
